// DAWNBlock_47296179863784
// MI455X (gfx1250) — compile-verified, matrix operand order checked by reading, not run on hardware
//
#include <hip/hip_runtime.h>

// ---------------------------------------------------------------------------
// Types for CDNA5 WMMA (wave32, 16x16x32 f16 -> f32)
// ---------------------------------------------------------------------------
typedef __attribute__((ext_vector_type(16))) _Float16 v16h;
typedef __attribute__((ext_vector_type(8)))  _Float16 v8h;
typedef __attribute__((ext_vector_type(8)))  float    v8f;

#define WMMA_F16(a, b, c) \
    __builtin_amdgcn_wmma_f32_16x16x32_f16(false, (a), false, (b), (short)0, (c), false, false)

// Load a 16x32 A/B fragment (NT layout: operand rows have contiguous K).
// p points at the lane's row start + kk.  half8 = (lane>=16)?8:0.
// Lane<16 holds K {0..7,16..23}; lane>=16 holds K {8..15,24..31}. 16B-aligned.
__device__ __forceinline__ v16h load_frag(const _Float16* p, int half8) {
    v8h lo = *(const v8h*)(p + half8);
    v8h hi = *(const v8h*)(p + 16 + half8);
    return __builtin_shufflevector(lo, hi, 0,1,2,3,4,5,6,7,8,9,10,11,12,13,14,15);
}

__device__ __forceinline__ float gelu_exact(float x) {
    return 0.5f * x * (1.0f + erff(x * 0.70710678118654752f));
}

// CDNA5 async global->LDS copy (16B per lane), tracked on ASYNCcnt.
// lds_dst: generic pointer into __shared__; low 32 bits are the wave-relative
// LDS byte address (flat-aperture mapping), which is what VDST supplies.
__device__ __forceinline__ void async_copy_b128(void* lds_dst, const void* gsrc) {
    unsigned ldso = (unsigned)(size_t)lds_dst;
    asm volatile("global_load_async_to_lds_b128 %0, %1, off"
                 :: "v"(ldso), "v"(gsrc) : "memory");
}
__device__ __forceinline__ void wait_async0() {
    asm volatile("s_wait_asynccnt 0x0" ::: "memory");
}

// ---------------------------------------------------------------------------
// Generic batched NT GEMM:  C[b] = epilogue( A[b] (MxK) * B[b]^T (NxK) )
// Block = 128 threads (4 waves); wave w computes rows [by*64+16w, +16), cols
// [bx*64, +64) as 4 WMMA accumulators.  K % 32 == 0, M % 64 == 0, N % 64 == 0.
// Register double-buffered k-loop: fragments for k+32 are in flight while the
// WMMAs for k execute, so s_wait_loadcnt tolerates outstanding loads.
// ---------------------------------------------------------------------------
__global__ void __launch_bounds__(128)
gemm_nt_wmma(const _Float16* __restrict__ A, long long sA,
             const _Float16* __restrict__ B, long long sB,
             const float* __restrict__ bias,
             const float* __restrict__ resid, long long sR,
             float* __restrict__ Cf, _Float16* __restrict__ Ch, long long sC,
             int N, int K, int gelu)
{
    const int bz   = blockIdx.z;
    const int lane = threadIdx.x & 31;
    const int wave = threadIdx.x >> 5;
    const int m0   = blockIdx.y * 64 + wave * 16;
    const int n0   = blockIdx.x * 64;
    const int lr   = lane & 15;
    const int half8 = (lane >> 4) * 8;

    const _Float16* Ab = A + (size_t)bz * sA;
    const _Float16* Bb = B + (size_t)bz * sB;
    const _Float16* arow = Ab + (size_t)(m0 + lr) * K;
    const _Float16* brow[4];
#pragma unroll
    for (int t = 0; t < 4; ++t) brow[t] = Bb + (size_t)(n0 + t * 16 + lr) * K;

    v8f acc[4] = {};

    // Prologue: fragments for kk = 0.
    v16h a_cur = load_frag(arow, half8);
    v16h b_cur[4];
#pragma unroll
    for (int t = 0; t < 4; ++t) b_cur[t] = load_frag(brow[t], half8);

#pragma unroll 2
    for (int kk = 0; kk < K; kk += 32) {
        const int kn = kk + 32;
        v16h a_nxt = a_cur;
        v16h b_nxt[4];
#pragma unroll
        for (int t = 0; t < 4; ++t) b_nxt[t] = b_cur[t];
        if (kn < K) {
            __builtin_prefetch(arow + kn + 96, 0, 1);   // global_prefetch_b8
            a_nxt = load_frag(arow + kn, half8);
#pragma unroll
            for (int t = 0; t < 4; ++t) b_nxt[t] = load_frag(brow[t] + kn, half8);
        }
#pragma unroll
        for (int t = 0; t < 4; ++t) acc[t] = WMMA_F16(a_cur, b_cur[t], acc[t]);
        a_cur = a_nxt;
#pragma unroll
        for (int t = 0; t < 4; ++t) b_cur[t] = b_nxt[t];
    }

    // Epilogue.  C-fragment layout: element j -> row m0+half8+j, col n0+16t+lr.
#pragma unroll
    for (int t = 0; t < 4; ++t) {
        const int col = n0 + t * 16 + lr;
        float cb = bias ? bias[col] : 0.0f;
#pragma unroll
        for (int j = 0; j < 8; ++j) {
            const int row = m0 + half8 + j;
            float v = acc[t][j] + cb;
            if (gelu) v = gelu_exact(v);
            if (resid) v += resid[(size_t)bz * sR + (size_t)row * N + col];
            const size_t o = (size_t)bz * sC + (size_t)row * N + col;
            if (Cf) Cf[o] = v;
            if (Ch) Ch[o] = (_Float16)v;
        }
    }
}

// ---------------------------------------------------------------------------
// Flash attention (non-causal), WMMA QK^T and PV, online softmax.
// qkv layout [B,S,3E] f16, heads of size DH.  Block = 128 thr (4 waves),
// wave handles 16 query rows; block shares 32-key K/V tiles staged in LDS by
// CDNA5 async global->LDS copies (ASYNCcnt), consumed via ds_load fragments.
// ---------------------------------------------------------------------------
template <int DH>
__global__ void __launch_bounds__(128)
flash_attn(const _Float16* __restrict__ qkv, _Float16* __restrict__ out,
           int S, int E)
{
    const int b    = blockIdx.z;
    const int h    = blockIdx.y;
    const int lane = threadIdx.x & 31;
    const int wave = threadIdx.x >> 5;
    const int q0   = blockIdx.x * 64 + wave * 16;
    const int lr   = lane & 15;
    const int half8 = (lane >> 4) * 8;
    const int tok  = 3 * E;

    __shared__ _Float16 Klds[32][DH];
    __shared__ _Float16 Vlds[32][DH];
    __shared__ _Float16 Plds[4][16][32];

    const _Float16* base = qkv + (size_t)b * S * tok;
    const float scale = 1.0f / sqrtf((float)DH);

    // Preload Q fragments for this wave's 16 rows.
    v16h qf[DH / 32];
    {
        const _Float16* qrow = base + (size_t)(q0 + lr) * tok + h * DH;
#pragma unroll
        for (int kk = 0; kk < DH; kk += 32) qf[kk / 32] = load_frag(qrow + kk, half8);
    }

    v8f acc[DH / 16] = {};
    float mrow[8], lrow[8];
#pragma unroll
    for (int j = 0; j < 8; ++j) { mrow[j] = -1e30f; lrow[j] = 0.0f; }

    for (int kt = 0; kt < S; kt += 32) {
        // Async-stage K/V tile (32 keys x DH halfs): 16B per lane per issue.
#pragma unroll
        for (int i = threadIdx.x; i < 4 * DH; i += 128) {
            const int r = (i * 8) / DH, c = (i * 8) % DH;
            const _Float16* gk = base + (size_t)(kt + r) * tok + E + h * DH + c;
            async_copy_b128(&Klds[r][c], gk);
            async_copy_b128(&Vlds[r][c], gk + E);
        }
        wait_async0();
        __syncthreads();

        // S_tile[16q, 32k] = Q * K^T  (two 16x16 C tiles).
        v8f s0 = {}, s1 = {};
#pragma unroll
        for (int kk = 0; kk < DH; kk += 32) {
            v8h k0 = *(const v8h*)&Klds[lr][kk + half8];
            v8h k1 = *(const v8h*)&Klds[lr][kk + 16 + half8];
            v16h b0 = __builtin_shufflevector(k0, k1, 0,1,2,3,4,5,6,7,8,9,10,11,12,13,14,15);
            v8h k2 = *(const v8h*)&Klds[16 + lr][kk + half8];
            v8h k3 = *(const v8h*)&Klds[16 + lr][kk + 16 + half8];
            v16h b1 = __builtin_shufflevector(k2, k3, 0,1,2,3,4,5,6,7,8,9,10,11,12,13,14,15);
            s0 = WMMA_F16(qf[kk / 32], b0, s0);
            s1 = WMMA_F16(qf[kk / 32], b1, s1);
        }

        // Online softmax update (row j+half8; 16-lane row reductions).
#pragma unroll
        for (int j = 0; j < 8; ++j) {
            float v0 = s0[j] * scale, v1 = s1[j] * scale;
            float mt = fmaxf(v0, v1);
            mt = fmaxf(mt, __shfl_xor(mt, 1));
            mt = fmaxf(mt, __shfl_xor(mt, 2));
            mt = fmaxf(mt, __shfl_xor(mt, 4));
            mt = fmaxf(mt, __shfl_xor(mt, 8));
            const float mnew  = fmaxf(mrow[j], mt);
            const float alpha = __expf(mrow[j] - mnew);
            const float p0 = __expf(v0 - mnew), p1 = __expf(v1 - mnew);
            float rs = p0 + p1;
            rs += __shfl_xor(rs, 1);
            rs += __shfl_xor(rs, 2);
            rs += __shfl_xor(rs, 4);
            rs += __shfl_xor(rs, 8);
            lrow[j] = lrow[j] * alpha + rs;
            mrow[j] = mnew;
#pragma unroll
            for (int t = 0; t < DH / 16; ++t) acc[t][j] *= alpha;
            Plds[wave][half8 + j][lr]      = (_Float16)p0;
            Plds[wave][half8 + j][16 + lr] = (_Float16)p1;
        }
        // Within-wave LDS bounce: DS ops are in-order; make the RAW explicit.
        asm volatile("s_wait_dscnt 0x0" ::: "memory");

        // Re-read P in A-fragment layout.
        v8h p0v = *(const v8h*)&Plds[wave][lr][half8];
        v8h p1v = *(const v8h*)&Plds[wave][lr][16 + half8];
        v16h pf = __builtin_shufflevector(p0v, p1v, 0,1,2,3,4,5,6,7,8,9,10,11,12,13,14,15);

        // acc += P * V.
#pragma unroll
        for (int t = 0; t < DH / 16; ++t) {
            v16h vf;
#pragma unroll
            for (int j = 0; j < 8; ++j) {
                vf[j]     = Vlds[half8 + j][t * 16 + lr];
                vf[8 + j] = Vlds[16 + half8 + j][t * 16 + lr];
            }
            acc[t] = WMMA_F16(pf, vf, acc[t]);
        }
        __syncthreads();
    }

    // Normalize and write [B,S,E] f16.
#pragma unroll
    for (int t = 0; t < DH / 16; ++t)
#pragma unroll
        for (int j = 0; j < 8; ++j) {
            const int row = q0 + half8 + j;
            out[((size_t)b * S + row) * E + h * DH + t * 16 + lr] =
                (_Float16)(acc[t][j] / lrow[j]);
        }
}

// ---------------------------------------------------------------------------
// Small helper kernels
// ---------------------------------------------------------------------------
__global__ void f32_to_f16_k(const float* __restrict__ in, _Float16* __restrict__ out, size_t n) {
    size_t i = (size_t)blockIdx.x * blockDim.x + threadIdx.x;
    const size_t stride = (size_t)gridDim.x * blockDim.x;
    for (; i < n; i += stride) out[i] = (_Float16)in[i];
}

// out[b,n] = max_s in[b,s,n]
__global__ void colmax_k(const float* __restrict__ in, float* __restrict__ out, int S, int N) {
    const int idx = blockIdx.x * blockDim.x + threadIdx.x;  // over B*N
    const int b = idx / N, n = idx % N;
    const float* p = in + (size_t)b * S * N + n;
    float m = -1e30f;
    for (int s = 0; s < S; ++s) m = fmaxf(m, p[(size_t)s * N]);
    out[idx] = m;
}

// One block per batch; blockDim == N_INPUT (256).  Rank-select top-k, softmax
// with temperature 2.0, renormalized weights at selected slots.
__global__ void __launch_bounds__(256)
router_topk_k(const float* __restrict__ scores, int* __restrict__ idx,
              float* __restrict__ selw, int Ninp, int k)
{
    const int b = blockIdx.x, t = threadIdx.x;
    __shared__ float lg[256];
    __shared__ float red[256];
    const float v = scores[b * Ninp + t] * 0.5f;   // / TEMP
    lg[t] = v;
    red[t] = v;
    __syncthreads();
    for (int s = 128; s > 0; s >>= 1) { if (t < s) red[t] = fmaxf(red[t], red[t + s]); __syncthreads(); }
    const float mx = red[0];
    __syncthreads();
    const float e = __expf(v - mx);
    red[t] = e;
    __syncthreads();
    for (int s = 128; s > 0; s >>= 1) { if (t < s) red[t] += red[t + s]; __syncthreads(); }
    const float soft = e / red[0];
    __syncthreads();
    int rank = 0;
    for (int j = 0; j < Ninp; ++j) {
        const float o = lg[j];
        rank += (o > v) || (o == v && j < t);
    }
    red[t] = (rank < k) ? soft : 0.0f;
    __syncthreads();
    for (int s = 128; s > 0; s >>= 1) { if (t < s) red[t] += red[t + s]; __syncthreads(); }
    const float ssum = red[0];
    if (rank < k) {
        idx[b * k + rank]  = t;
        selw[b * k + rank] = soft / (ssum + 1e-8f);
    }
}

// relevance[b,p] = a2( gelu( a1(input_sel) ) )  -- one block per batch.
__global__ void __launch_bounds__(512)
relevance_k(const int* __restrict__ idx, const float* __restrict__ a1w,
            const float* __restrict__ a1b, const float* __restrict__ a2w,
            const float* __restrict__ a2b, float* __restrict__ rel, int k)
{
    const int b = blockIdx.x, t = threadIdx.x;
    __shared__ float h[512];
    float s = a1b[t];
    for (int j = 0; j < k; ++j) s += a1w[t * 256 + idx[b * k + j]];
    h[t] = gelu_exact(s);
    __syncthreads();
    float r = a2b[t];
    for (int i = 0; i < 512; ++i) r += a2w[t * 512 + i] * h[i];
    rel[b * 512 + t] = r;
}

__global__ void __launch_bounds__(512)
proc_topk_k(const float* __restrict__ ascore, const float* __restrict__ rel,
            int* __restrict__ pidx, int P, int k)
{
    const int b = blockIdx.x, t = threadIdx.x;
    __shared__ float fs[512];
    const float v = ascore[b * P + t] * (1.0f / (1.0f + __expf(-rel[b * P + t])));
    fs[t] = v;
    __syncthreads();
    int rank = 0;
    for (int j = 0; j < P; ++j) {
        const float o = fs[j];
        rank += (o > v) || (o == v && j < t);
    }
    if (rank < k) pidx[b * k + rank] = t;
}

// LayerNorm over last dim (256); one block per row.
__global__ void __launch_bounds__(256)
layernorm_k(const float* __restrict__ in, const float* __restrict__ g,
            const float* __restrict__ bb, float* __restrict__ out)
{
    __shared__ float sh[256];
    const int row = blockIdx.x, t = threadIdx.x;
    const float v = in[(size_t)row * 256 + t];
    sh[t] = v;
    __syncthreads();
    for (int s = 128; s > 0; s >>= 1) { if (t < s) sh[t] += sh[t + s]; __syncthreads(); }
    const float mean = sh[0] * (1.0f / 256.0f);
    __syncthreads();
    const float d = v - mean;
    sh[t] = d * d;
    __syncthreads();
    for (int s = 128; s > 0; s >>= 1) { if (t < s) sh[t] += sh[t + s]; __syncthreads(); }
    const float var = sh[0] * (1.0f / 256.0f);
    out[(size_t)row * 256 + t] = d * rsqrtf(var + 1e-5f) * g[t] + bb[t];
}

// Gathers (produce f16 WMMA operands)
__global__ void gather_acts_k(const float* __restrict__ acts, const int* __restrict__ idx,
                              const float* __restrict__ selw, _Float16* __restrict__ Ag,
                              int Ninp, int k)
{
    const int b = blockIdx.y, s = blockIdx.x, j = threadIdx.x;
    const size_t row = (size_t)b * gridDim.x + s;
    Ag[row * k + j] = (_Float16)(acts[row * Ninp + idx[b * k + j]] * selw[b * k + j]);
}

__global__ void gather_cw_k(const float* __restrict__ comb, const int* __restrict__ idx,
                            _Float16* __restrict__ Bg, int Ninp, int k)
{
    const int b = blockIdx.y, p = blockIdx.x, j = threadIdx.x;
    Bg[((size_t)b * gridDim.x + p) * k + j] = (_Float16)comb[p * Ninp + idx[b * k + j]];
}

__global__ void gather_proc_k(const float* __restrict__ pa, const int* __restrict__ pidx,
                              _Float16* __restrict__ Ag2, int P, int k)
{
    const int b = blockIdx.y, s = blockIdx.x, j = threadIdx.x;
    const size_t row = (size_t)b * gridDim.x + s;
    Ag2[row * k + j] = (_Float16)pa[row * P + pidx[b * k + j]];
}

__global__ void gather_wT_k(const float* __restrict__ wout, const int* __restrict__ pidx,
                            _Float16* __restrict__ Wg, int D, int k)
{
    const int b = blockIdx.y, d = blockIdx.x, j = threadIdx.x;
    Wg[((size_t)b * D + d) * k + j] = (_Float16)wout[(size_t)pidx[b * k + j] * D + d];
}

// ---------------------------------------------------------------------------
// Orchestration
// ---------------------------------------------------------------------------
static constexpr int Bsz = 8, S = 1024, D = 1024;
static constexpr int NI = 256, NP = 512, HID = 512;
static constexpr int KIN = 128, KPR = 256;

extern "C" void kernel_launch(void* const* d_in, const int* in_sizes, int n_in,
                              void* d_out, int out_size, void* d_ws, size_t ws_size,
                              hipStream_t stream)
{
    const float* x      = (const float*)d_in[0];
    const float* rinw   = (const float*)d_in[1];
    const float* rinb   = (const float*)d_in[2];
    const float* routw  = (const float*)d_in[3];
    const float* routb  = (const float*)d_in[4];
    const float* affw   = (const float*)d_in[5];
    const float* affb   = (const float*)d_in[6];
    const float* pat    = (const float*)d_in[7];
    const float* iinw   = (const float*)d_in[8];
    const float* iinb   = (const float*)d_in[9];
    const float* ioutw  = (const float*)d_in[10];
    const float* ioutb  = (const float*)d_in[11];
    const float* lng    = (const float*)d_in[12];
    const float* lnb    = (const float*)d_in[13];
    const float* combw  = (const float*)d_in[14];
    const float* outpw  = (const float*)d_in[15];
    const float* a1w    = (const float*)d_in[16];
    const float* a1b    = (const float*)d_in[17];
    const float* a2w    = (const float*)d_in[18];
    const float* a2b    = (const float*)d_in[19];
    float* out = (float*)d_out;

    // Bump-allocate workspace (256B aligned).
    char* wp = (char*)d_ws;
    auto alloc = [&](size_t bytes) -> void* {
        void* p = wp;
        wp += (bytes + 255) & ~(size_t)255;
        return p;
    };
    const size_t BS = (size_t)Bsz * S;                       // 8192 rows
    _Float16* x16    = (_Float16*)alloc(BS * D * 2);
    _Float16* rinw16 = (_Float16*)alloc((size_t)3 * D * D * 2);
    _Float16* routw16= (_Float16*)alloc((size_t)D * D * 2);
    _Float16* affw16 = (_Float16*)alloc((size_t)NI * D * 2);
    _Float16* pat16  = (_Float16*)alloc((size_t)NI * D * 2);
    _Float16* iinw16 = (_Float16*)alloc((size_t)3 * NI * NI * 2);
    _Float16* ioutw16= (_Float16*)alloc((size_t)NI * NI * 2);
    _Float16* qkv1   = (_Float16*)alloc(BS * 3 * D * 2);
    _Float16* attn1  = (_Float16*)alloc(BS * D * 2);
    _Float16* ctx16  = (_Float16*)alloc(BS * D * 2);
    float*    affbuf = (float*)   alloc(BS * NI * 4);
    float*    actsf  = (float*)   alloc(BS * NI * 4);
    _Float16* acts16 = (_Float16*)alloc(BS * NI * 2);
    _Float16* qkv2   = (_Float16*)alloc(BS * 3 * NI * 2);
    _Float16* attn2  = (_Float16*)alloc(BS * NI * 2);
    float*    preln  = (float*)   alloc(BS * NI * 4);
    float*    acts2  = (float*)   alloc(BS * NI * 4);
    float*    scores = (float*)   alloc((size_t)Bsz * NI * 4);
    int*      idxb   = (int*)     alloc((size_t)Bsz * KIN * 4);
    float*    selw   = (float*)   alloc((size_t)Bsz * KIN * 4);
    _Float16* Ag     = (_Float16*)alloc(BS * KIN * 2);
    _Float16* Bg     = (_Float16*)alloc((size_t)Bsz * NP * KIN * 2);
    float*    pacts  = (float*)   alloc(BS * NP * 4);
    float*    ascore = (float*)   alloc((size_t)Bsz * NP * 4);
    float*    rel    = (float*)   alloc((size_t)Bsz * NP * 4);
    int*      pidxb  = (int*)     alloc((size_t)Bsz * KPR * 4);
    _Float16* Ag2    = (_Float16*)alloc(BS * KPR * 2);
    _Float16* Wg     = (_Float16*)alloc((size_t)Bsz * D * KPR * 2);

    auto cvt = [&](const float* src, _Float16* dst, size_t n) {
        int blocks = (int)((n + 255) / 256);
        if (blocks > 8192) blocks = 8192;
        f32_to_f16_k<<<blocks, 256, 0, stream>>>(src, dst, n);
    };
    cvt(x,     x16,     BS * D);
    cvt(rinw,  rinw16,  (size_t)3 * D * D);
    cvt(routw, routw16, (size_t)D * D);
    cvt(affw,  affw16,  (size_t)NI * D);
    cvt(pat,   pat16,   (size_t)NI * D);
    cvt(iinw,  iinw16,  (size_t)3 * NI * NI);
    cvt(ioutw, ioutw16, (size_t)NI * NI);

    // 1) Router QKV:  [8192,3072] = x16 * r_in_w^T + b
    gemm_nt_wmma<<<dim3(3 * D / 64, BS / 64, 1), 128, 0, stream>>>(
        x16, 0, rinw16, 0, rinb, nullptr, 0, nullptr, qkv1, 0, 3 * D, D, 0);

    // 2) Router flash attention (8 heads, dh=128)
    flash_attn<128><<<dim3(S / 64, 8, Bsz), 128, 0, stream>>>(qkv1, attn1, S, D);

    // 3) Router out-proj -> context (f16)
    gemm_nt_wmma<<<dim3(D / 64, BS / 64, 1), 128, 0, stream>>>(
        attn1, 0, routw16, 0, routb, nullptr, 0, nullptr, ctx16, 0, D, D, 0);

    // 4) affinity = context * aff_w^T + b   (f32)
    gemm_nt_wmma<<<dim3(NI / 64, BS / 64, 1), 128, 0, stream>>>(
        ctx16, 0, affw16, 0, affb, nullptr, 0, affbuf, nullptr, 0, NI, D, 0);

    // 5) acts = gelu(context * patterns^T)  (f32 + f16)
    gemm_nt_wmma<<<dim3(NI / 64, BS / 64, 1), 128, 0, stream>>>(
        ctx16, 0, pat16, 0, nullptr, nullptr, 0, actsf, acts16, 0, NI, D, 1);

    // 6) scores = max_s affinity;  router top-k + weights
    colmax_k<<<(Bsz * NI) / 256, 256, 0, stream>>>(affbuf, scores, S, NI);
    router_topk_k<<<Bsz, 256, 0, stream>>>(scores, idxb, selw, NI, KIN);

    // 7) InputNeurons MHA (4 heads, dh=64) + residual
    gemm_nt_wmma<<<dim3(3 * NI / 64, BS / 64, 1), 128, 0, stream>>>(
        acts16, 0, iinw16, 0, iinb, nullptr, 0, nullptr, qkv2, 0, 3 * NI, NI, 0);
    flash_attn<64><<<dim3(S / 64, 4, Bsz), 128, 0, stream>>>(qkv2, attn2, S, NI);
    gemm_nt_wmma<<<dim3(NI / 64, BS / 64, 1), 128, 0, stream>>>(
        attn2, 0, ioutw16, 0, ioutb, actsf, 0, preln, nullptr, 0, NI, NI, 0);
    layernorm_k<<<(int)BS, 256, 0, stream>>>(preln, lng, lnb, acts2);

    // 8) Gather routed neurons -> per-batch GEMM operands
    gather_acts_k<<<dim3(S, Bsz), KIN, 0, stream>>>(acts2, idxb, selw, Ag, NI, KIN);
    gather_cw_k<<<dim3(NP, Bsz), KIN, 0, stream>>>(combw, idxb, Bg, NI, KIN);

    // 9) proc_acts = gelu(sel_acts @ sel_cw^T)  [B] x [1024,512,128]
    gemm_nt_wmma<<<dim3(NP / 64, S / 64, Bsz), 128, 0, stream>>>(
        Ag, (long long)S * KIN, Bg, (long long)NP * KIN, nullptr, nullptr, 0,
        pacts, nullptr, (long long)S * NP, NP, KIN, 1);

    // 10) process scoring + top-k
    colmax_k<<<(Bsz * NP) / 256, 256, 0, stream>>>(pacts, ascore, S, NP);
    relevance_k<<<Bsz, HID, 0, stream>>>(idxb, a1w, a1b, a2w, a2b, rel, KIN);
    proc_topk_k<<<Bsz, NP, 0, stream>>>(ascore, rel, pidxb, NP, KPR);

    // 11) Final routed projection + residual into d_out
    gather_proc_k<<<dim3(S, Bsz), KPR, 0, stream>>>(pacts, pidxb, Ag2, NP, KPR);
    gather_wT_k<<<dim3(D, Bsz), KPR, 0, stream>>>(outpw, pidxb, Wg, D, KPR);
    gemm_nt_wmma<<<dim3(D / 64, S / 64, Bsz), 128, 0, stream>>>(
        Ag2, (long long)S * KPR, Wg, (long long)D * KPR, nullptr,
        x, (long long)S * D, out, nullptr, (long long)S * D, D, KPR, 0);
}
